// LSTMForecaster_45526653337975
// MI455X (gfx1250) — matrix operand order repaired and checked by reading, not compiled, not run on hardware
//
#include <hip/hip_runtime.h>
#include <hip/hip_bf16.h>

typedef __bf16 v16bf __attribute__((ext_vector_type(16)));
typedef __bf16 v8bf  __attribute__((ext_vector_type(8)));
typedef float  v8f   __attribute__((ext_vector_type(8)));

#define BB   2048
#define SEQ  512
#define HH   64
#define G4   256
#define NOUT 72
#define LDSP 72   // padded LDS row stride (bf16 elements) for h tiles

// ---------------------------------------------------------------------------
// Prep kernel: pack the three [256,64] f32 weight matrices into bf16 WMMA
// B-fragment order, and fold the bias pairs.
//   B fragment (matrix m, kblock kb, ntile nt), lane L holds 16 bf16:
//   N = nt*16 + (L%16);  K = kb*32 + (L<16 ? 0 : 8) + (e<8 ? e : 16 + e - 8),  e=0..15
// ---------------------------------------------------------------------------
__global__ void lstm_prepack_kernel(const float* __restrict__ Whh0,
                                    const float* __restrict__ Wih1,
                                    const float* __restrict__ Whh1,
                                    const float* __restrict__ bih0,
                                    const float* __restrict__ bhh0,
                                    const float* __restrict__ bih1,
                                    const float* __restrict__ bhh1,
                                    unsigned short* __restrict__ packed,
                                    float* __restrict__ bias0,
                                    float* __restrict__ bias1)
{
    int idx = blockIdx.x * blockDim.x + threadIdx.x;
    const int total = 3 * 2 * 16 * 32 * 16;   // 49152 bf16 elements
    if (idx < total) {
        int e    = idx & 15;
        int lane = (idx >> 4) & 31;
        int nt   = (idx >> 9) & 15;
        int kb   = (idx >> 13) & 1;
        int m    = idx >> 14;
        int n = nt * 16 + (lane & 15);
        int k = kb * 32 + ((lane < 16) ? 0 : 8) + ((e < 8) ? e : 16 + (e - 8));
        const float* W = (m == 0) ? Whh0 : (m == 1) ? Wih1 : Whh1;
        union { float f; unsigned u; } cv;
        cv.f = W[n * HH + k];
        unsigned b = cv.u + 0x7FFFu + ((cv.u >> 16) & 1u);   // f32 -> bf16 RNE
        packed[idx] = (unsigned short)(b >> 16);
    }
    if (idx < G4) {
        bias0[idx] = bih0[idx] + bhh0[idx];
        bias1[idx] = bih1[idx] + bhh1[idx];
    }
}

// ---------------------------------------------------------------------------
// Fast gate math. Prefer the CDNA5 native v_tanh_f32 trans op when the
// toolchain exposes it; otherwise fall back to v_exp + v_rcp.
// ---------------------------------------------------------------------------
#if defined(__has_builtin)
#if __has_builtin(__builtin_amdgcn_tanhf)
#define LSTM_HAS_TANH 1
#endif
#endif

__device__ __forceinline__ float rcp_(float x) { return __builtin_amdgcn_rcpf(x); }

#ifdef LSTM_HAS_TANH
__device__ __forceinline__ float tanh_(float x)    { return __builtin_amdgcn_tanhf(x); }
__device__ __forceinline__ float sigmoid_(float x) { return fmaf(0.5f, __builtin_amdgcn_tanhf(0.5f * x), 0.5f); }
#else
__device__ __forceinline__ float tanh_(float x)    { return fmaf(-2.0f, rcp_(__expf(2.0f * x) + 1.0f), 1.0f); }
__device__ __forceinline__ float sigmoid_(float x) { return rcp_(1.0f + __expf(-x)); }
#endif

__device__ __forceinline__ v16bf load_a_frag(const __bf16* lds, int row, int kst) {
    // 16-bit A-matrix 16x32 layout: lane holds K=[kst..kst+7] and [kst+16..kst+23]
    const v8bf lo = *(const v8bf*)(lds + row * LDSP + kst);
    const v8bf hi = *(const v8bf*)(lds + row * LDSP + kst + 16);
    v16bf r;
#pragma unroll
    for (int i = 0; i < 8; ++i) { r[i] = lo[i]; r[i + 8] = hi[i]; }
    return r;
}

__device__ __forceinline__ v8f lstm_combine(const v8f* acc, v8f& cst) {
    v8f h;
#pragma unroll
    for (int j = 0; j < 8; ++j) {
        float ig = sigmoid_(acc[0][j]);
        float fg = sigmoid_(acc[1][j]);
        float gg = tanh_(acc[2][j]);
        float og = sigmoid_(acc[3][j]);
        float c  = fmaf(fg, cst[j], ig * gg);
        cst[j] = c;
        h[j]   = og * tanh_(c);
    }
    return h;
}

// ---------------------------------------------------------------------------
// Fused 2-layer LSTM + FC, wave-specialized pipeline.
// Grid: 128 WGs x 256 threads (8 waves). WG owns 16 batch rows.
//   waves 0-3 : layer 0, iteration k produces h0[k]      (k = 0..S-1)
//   waves 4-7 : layer 1, iteration k produces h1[k-1]    (k = 1..S)
// h0/h1 double-buffered in LDS -> one barrier per iteration.
// ---------------------------------------------------------------------------
__global__ __launch_bounds__(256, 1)
void lstm2_pipelined_kernel(const float* __restrict__ x,        // [B,S,1]
                            const float* __restrict__ wih0,     // [256]
                            const unsigned short* __restrict__ packed,
                            const float* __restrict__ bias0,    // [256]
                            const float* __restrict__ bias1,    // [256]
                            const float* __restrict__ fc_w,     // [72,64]
                            const float* __restrict__ fc_b,     // [72]
                            float* __restrict__ out)            // [B,72]
{
    __shared__ __bf16 h0buf[2][16 * LDSP];
    __shared__ __bf16 h1buf[2][16 * LDSP];

    const int tid    = threadIdx.x;
    const int wave   = tid >> 5;
    const int lane   = tid & 31;
    const int sub    = wave & 3;          // h-column tile this wave owns
    const bool gA    = (wave < 4);        // layer-0 group
    const int l16    = lane & 15;
    const int hi8    = (lane < 16) ? 0 : 8;
    const int wgbase = blockIdx.x * 16;

    // ---- weight fragments: unioned storage, max 128 VGPRs ----
    const v16bf* pw = (const v16bf*)packed;
    v16bf wfrag[16];
    float wih0_n[4], bias_n[4];
    if (gA) {
#pragma unroll
        for (int g = 0; g < 4; ++g) {
            const int nt = g * 4 + sub;
#pragma unroll
            for (int kb = 0; kb < 2; ++kb)
                wfrag[g * 2 + kb] = pw[(((0 * 2 + kb) * 16) + nt) * 32 + lane];
            const int n = g * 64 + sub * 16 + l16;
            wih0_n[g] = wih0[n];
            bias_n[g] = bias0[n];
        }
    } else {
#pragma unroll
        for (int g = 0; g < 4; ++g) {
            const int nt = g * 4 + sub;
#pragma unroll
            for (int kb = 0; kb < 2; ++kb) {
                wfrag[g * 4 + kb]     = pw[(((1 * 2 + kb) * 16) + nt) * 32 + lane];
                wfrag[g * 4 + 2 + kb] = pw[(((2 * 2 + kb) * 16) + nt) * 32 + lane];
            }
            bias_n[g] = bias1[g * 64 + sub * 16 + l16];
        }
    }

    // x element offsets for the 8 batch rows this lane accumulates (layer-0 only)
    int xoff[8];
#pragma unroll
    for (int j = 0; j < 8; ++j) xoff[j] = (wgbase + hi8 + j) * SEQ;

    // zero-init both double buffers
    {
        __bf16* z0 = &h0buf[0][0];
        __bf16* z1 = &h1buf[0][0];
        for (int i = tid; i < 2 * 16 * LDSP; i += 256) {
            z0[i] = (__bf16)0.0f;
            z1[i] = (__bf16)0.0f;
        }
    }
    v8f cst;                              // c0 for group A, c1 for group B
#pragma unroll
    for (int j = 0; j < 8; ++j) cst[j] = 0.0f;

    const int Arow = l16;
    const int ksel = (lane < 16) ? 0 : 8;
    const int col  = sub * 16 + l16;

    for (int k = 0; k <= SEQ; ++k) {
        __syncthreads();

        if (gA) {
            // ---------------- layer 0: produce h0[k] ----------------
            if (k < SEQ) {
                const int wb = k & 1;
                v16bf a0[2];
#pragma unroll
                for (int kb = 0; kb < 2; ++kb)
                    a0[kb] = load_a_frag(&h0buf[wb ^ 1][0], Arow, kb * 32 + ksel);

                float xv[8];
#pragma unroll
                for (int j = 0; j < 8; ++j) xv[j] = x[xoff[j] + k];

                v8f acc[4];
#pragma unroll
                for (int g = 0; g < 4; ++g) {
                    v8f ci;
#pragma unroll
                    for (int j = 0; j < 8; ++j) ci[j] = fmaf(xv[j], wih0_n[g], bias_n[g]);
                    ci     = __builtin_amdgcn_wmma_f32_16x16x32_bf16(false, a0[0], false, wfrag[g * 2 + 0], (short)0, ci, false, false);
                    acc[g] = __builtin_amdgcn_wmma_f32_16x16x32_bf16(false, a0[1], false, wfrag[g * 2 + 1], (short)0, ci, false, false);
                }

                v8f h0n = lstm_combine(acc, cst);
#pragma unroll
                for (int j = 0; j < 8; ++j)
                    h0buf[wb][(hi8 + j) * LDSP + col] = (__bf16)h0n[j];
            }
        } else {
            // ---------------- layer 1: produce h1[k-1] ----------------
            if (k >= 1) {
                const int t  = k - 1;
                const int wb = t & 1;
                v16bf a0[2], a1[2];
#pragma unroll
                for (int kb = 0; kb < 2; ++kb) {
                    a0[kb] = load_a_frag(&h0buf[wb][0],     Arow, kb * 32 + ksel); // h0[t]
                    a1[kb] = load_a_frag(&h1buf[wb ^ 1][0], Arow, kb * 32 + ksel); // h1[t-1]
                }

                v8f acc[4];
#pragma unroll
                for (int g = 0; g < 4; ++g) {
                    v8f ci;
#pragma unroll
                    for (int j = 0; j < 8; ++j) ci[j] = bias_n[g];
                    ci     = __builtin_amdgcn_wmma_f32_16x16x32_bf16(false, a0[0], false, wfrag[g * 4 + 0], (short)0, ci, false, false);
                    ci     = __builtin_amdgcn_wmma_f32_16x16x32_bf16(false, a0[1], false, wfrag[g * 4 + 1], (short)0, ci, false, false);
                    ci     = __builtin_amdgcn_wmma_f32_16x16x32_bf16(false, a1[0], false, wfrag[g * 4 + 2], (short)0, ci, false, false);
                    acc[g] = __builtin_amdgcn_wmma_f32_16x16x32_bf16(false, a1[1], false, wfrag[g * 4 + 3], (short)0, ci, false, false);
                }

                v8f h1n = lstm_combine(acc, cst);
#pragma unroll
                for (int j = 0; j < 8; ++j)
                    h1buf[wb][(hi8 + j) * LDSP + col] = (__bf16)h1n[j];
            }
        }
    }

    __syncthreads();

    // ---- final FC: out[b,:] = h1[S-1][b,:] @ fc_w^T + fc_b ----
    const __bf16* hf = &h1buf[(SEQ - 1) & 1][0];
    for (int p = tid; p < 16 * NOUT; p += 256) {
        const int r = p / NOUT;
        const int o = p % NOUT;
        float s = fc_b[o];
#pragma unroll
        for (int kk = 0; kk < HH; ++kk)
            s += (float)hf[r * LDSP + kk] * fc_w[o * HH + kk];
        out[(wgbase + r) * NOUT + o] = s;
    }
}

extern "C" void kernel_launch(void* const* d_in, const int* in_sizes, int n_in,
                              void* d_out, int out_size, void* d_ws, size_t ws_size,
                              hipStream_t stream) {
    (void)in_sizes; (void)n_in; (void)out_size; (void)ws_size;
    const float* x       = (const float*)d_in[0];
    const float* W_ih_l0 = (const float*)d_in[1];   // [256,1]
    const float* W_hh_l0 = (const float*)d_in[2];   // [256,64]
    const float* b_ih_l0 = (const float*)d_in[3];
    const float* b_hh_l0 = (const float*)d_in[4];
    const float* W_ih_l1 = (const float*)d_in[5];   // [256,64]
    const float* W_hh_l1 = (const float*)d_in[6];   // [256,64]
    const float* b_ih_l1 = (const float*)d_in[7];
    const float* b_hh_l1 = (const float*)d_in[8];
    const float* fc_w    = (const float*)d_in[9];   // [72,64]
    const float* fc_b    = (const float*)d_in[10];  // [72]

    unsigned short* packed = (unsigned short*)d_ws;             // 49152 bf16
    float* bias0 = (float*)((char*)d_ws + 49152 * sizeof(unsigned short));
    float* bias1 = bias0 + G4;

    lstm_prepack_kernel<<<(49152 + 255) / 256, 256, 0, stream>>>(
        W_hh_l0, W_ih_l1, W_hh_l1, b_ih_l0, b_hh_l0, b_ih_l1, b_hh_l1,
        packed, bias0, bias1);

    lstm2_pipelined_kernel<<<BB / 16, 256, 0, stream>>>(
        x, W_ih_l0, packed, bias0, bias1, fc_w, fc_b, (float*)d_out);
}
